// MSDeformAttnTransformerEncoderOnly_58841051955634
// MI455X (gfx1250) — matrix operand order repaired and checked by reading, not compiled, not run on hardware
//
#include <hip/hip_runtime.h>
#include <hip/hip_bf16.h>
#include <stdint.h>

// ---------------- problem constants ----------------
#define B_      2
#define D_      256
#define NH_     8
#define NL_     4
#define NP_     4
#define DFF_    1024
#define NLAYERS_ 6
#define DH_     32
#define Q_      21760          // 128*128 + 64*64 + 32*32 + 16*16
#define M_      (B_ * Q_)      // 43520 rows (multiple of 128)
#define NOA_    384            // fused offsets(256) + attn-logits(128)

typedef __attribute__((ext_vector_type(16))) _Float16 v16h;
typedef __attribute__((ext_vector_type(8)))  float    v8f;

// ============================================================
// Pack [B,D,H,W] level tensors into token-major x/pos [B,Q,D]
// ============================================================
__global__ void pack_inputs(const float* __restrict__ s0, const float* __restrict__ p0,
                            const float* __restrict__ s1, const float* __restrict__ p1,
                            const float* __restrict__ s2, const float* __restrict__ p2,
                            const float* __restrict__ s3, const float* __restrict__ p3,
                            const float* __restrict__ lev,
                            float* __restrict__ x, float* __restrict__ pos,
                            _Float16* __restrict__ xh)
{
    int i = blockIdx.x * blockDim.x + threadIdx.x;
    const int total = M_ * D_;
    if (i >= total) return;
    int d = i % D_;
    int q = (i / D_) % Q_;
    int b = i / (D_ * Q_);
    int lvl, start, hw;
    const float *sp, *pp;
    if (q < 16384)      { lvl = 0; start = 0;     hw = 16384; sp = s0; pp = p0; }
    else if (q < 20480) { lvl = 1; start = 16384; hw = 4096;  sp = s1; pp = p1; }
    else if (q < 21504) { lvl = 2; start = 20480; hw = 1024;  sp = s2; pp = p2; }
    else                { lvl = 3; start = 21504; hw = 256;   sp = s3; pp = p3; }
    int p = q - start;
    size_t src_idx = (size_t)(b * D_ + d) * hw + p;
    float xv = sp[src_idx];
    float pv = pp[src_idx] + lev[lvl * D_ + d];
    x[i]   = xv;
    pos[i] = pv;
    xh[i]  = (_Float16)xv;
}

// ============================================================
// Weight convert+transpose into (possibly fused) N-major f16 buffers:
// W [NLAYERS][K][N] f32 -> Wt[l*dstLayerElems + (rowOff+n)*K + k] f16
// ============================================================
__global__ void convert_wt(const float* __restrict__ W, _Float16* __restrict__ Wt,
                           int K, int N, long dstLayerElems, int rowOff)
{
    long i = (long)blockIdx.x * blockDim.x + threadIdx.x;
    long per = (long)K * N;
    long total = (long)NLAYERS_ * per;
    if (i >= total) return;
    int  l = (int)(i / per);
    long r = i - (long)l * per;
    int  n = (int)(r / K);
    int  k = (int)(r % K);
    Wt[(long)l * dstLayerElems + (long)(rowOff + n) * K + k] =
        (_Float16)W[(long)l * per + (long)k * N + n];
}

// ============================================================
// Fused bias for the offsets+logits GEMM: [l][384] = [b_off | b_aw]
// ============================================================
__global__ void concat_bias(const float* __restrict__ boff, const float* __restrict__ baw,
                            float* __restrict__ bc)
{
    int i = blockIdx.x * blockDim.x + threadIdx.x;
    if (i >= NLAYERS_ * NOA_) return;
    int l = i / NOA_;
    int n = i % NOA_;
    bc[i] = (n < 256) ? boff[l * 256 + n] : baw[l * 128 + (n - 256)];
}

// ============================================================
// q = f16(x + pos)
// ============================================================
__global__ void add_to_f16(const float* __restrict__ a, const float* __restrict__ b,
                           _Float16* __restrict__ o, int total)
{
    int i = blockIdx.x * blockDim.x + threadIdx.x;
    if (i < total) o[i] = (_Float16)(a[i] + b[i]);
}

// ============================================================
// WMMA GEMM: C[M,N] = A[M,K](f16) * Bt[N,K](f16)^T + bias,
// optional relu, optional f32 and/or f16 outputs.
// Block = 256 threads = 8 waves (4 in M x 2 in N).
// Wave tile = 32M x 64N (2 A-frags x 4 B-frags -> 8 WMMA / K-step).
// Register double-buffered over K: loads for step k+32 issued before
// the WMMAs consuming step k (hides load latency behind matrix ops).
// Requires M%128==0, N%128==0, K%64==0.
// ============================================================
union FragU { v16h v; uint32_t u[8]; };

__global__ void __launch_bounds__(256) gemm_wmma_f16(
    const _Float16* __restrict__ A, const _Float16* __restrict__ Bt,
    const float* __restrict__ bias,
    float* __restrict__ Cf, _Float16* __restrict__ Ch,
    int M, int N, int K, int relu)
{
    const int lane = threadIdx.x & 31;
    const int wave = threadIdx.x >> 5;
    const int wm = wave & 3;
    const int wn = wave >> 2;
    const int row0 = blockIdx.x * 128 + wm * 32;
    const int col0 = blockIdx.y * 128 + wn * 64;
    const int h  = lane >> 4;     // lane half selects K octet
    const int ml = lane & 15;     // M (for A) / N (for B) within tile

    // A fragment dword->K offsets (ISA 16-bit A 16x32 layout)
    int ka[8];
#pragma unroll
    for (int j = 0; j < 8; ++j)
        ka[j] = (j < 4) ? (8 * h + 2 * j) : (16 + 8 * h + 2 * (j - 4));
    // B fragment dword->K offsets (N-major Bt, same K order as A)
    int kb[8];
#pragma unroll
    for (int j = 0; j < 8; ++j)
        kb[j] = (j < 4) ? (8 * h + 2 * j) : (16 + 8 * h + 2 * (j - 4));

    const _Float16* Ar[2];
    Ar[0] = A + (size_t)(row0 + ml) * K;
    Ar[1] = A + (size_t)(row0 + 16 + ml) * K;
    const _Float16* Bp[4];
#pragma unroll
    for (int t = 0; t < 4; ++t)
        Bp[t] = Bt + (size_t)(col0 + t * 16 + ml) * K;

    v8f acc[2][4] = {};

    FragU a0[2], b0[4], a1[2], b1[4];

    auto loadfr = [&](FragU (&af)[2], FragU (&bf)[4], int kk) {
#pragma unroll
        for (int mt = 0; mt < 2; ++mt)
#pragma unroll
            for (int j = 0; j < 8; ++j)
                af[mt].u[j] = *(const uint32_t*)(Ar[mt] + kk + ka[j]);
#pragma unroll
        for (int t = 0; t < 4; ++t)
#pragma unroll
            for (int j = 0; j < 8; ++j)
                bf[t].u[j] = *(const uint32_t*)(Bp[t] + kk + kb[j]);
    };
    auto domma = [&](FragU (&af)[2], FragU (&bf)[4]) {
#pragma unroll
        for (int mt = 0; mt < 2; ++mt)
#pragma unroll
            for (int t = 0; t < 4; ++t)
                acc[mt][t] = __builtin_amdgcn_wmma_f32_16x16x32_f16(
                    false, af[mt].v, false, bf[t].v, (short)0, acc[mt][t], false, false);
    };

    loadfr(a0, b0, 0);
    int kk = 0;
    for (; kk + 64 < K; kk += 64) {
        loadfr(a1, b1, kk + 32);
        domma(a0, b0);
        loadfr(a0, b0, kk + 64);
        domma(a1, b1);
    }
    loadfr(a1, b1, kk + 32);
    domma(a0, b0);
    domma(a1, b1);

    // Epilogue: C/D layout -> row = row0 + mt*16 + r + 8h, col = col0 + t*16 + ml
#pragma unroll
    for (int t = 0; t < 4; ++t) {
        int gcol = col0 + t * 16 + ml;
        float bv = bias[gcol];
#pragma unroll
        for (int mt = 0; mt < 2; ++mt) {
#pragma unroll
            for (int r = 0; r < 8; ++r) {
                int grow = row0 + mt * 16 + r + 8 * h;
                float val = acc[mt][t][r] + bv;
                if (relu) val = fmaxf(val, 0.0f);
                size_t idx = (size_t)grow * N + gcol;
                if (Cf) Cf[idx] = val;
                if (Ch) Ch[idx] = (_Float16)val;
            }
        }
    }
}

// ============================================================
// Deformable sampling + softmax + weighted sum.
// One wave per (b,q,head); lane = channel (DH==32==wave32).
// Reads fused offaw [M,384]: cols 0..255 offsets, 256..383 logits.
// ============================================================
__global__ void __launch_bounds__(128) deform_sample(
    const float* __restrict__ offaw, const float* __restrict__ v,
    _Float16* __restrict__ attn)
{
    const int starts[4] = {0, 16384, 20480, 21504};
    const int widths[4] = {128, 64, 32, 16};
    int lane = threadIdx.x & 31;
    int wid  = blockIdx.x * 4 + (threadIdx.x >> 5);   // global wave id in [0, M*NH)
    int hh = wid % NH_;
    int q  = (wid / NH_) % Q_;
    int b  = wid / (NH_ * Q_);

    // reference point (same for all levels; valid_ratios == 1)
    int lq;
    if (q < 16384) lq = 0; else if (q < 20480) lq = 1; else if (q < 21504) lq = 2; else lq = 3;
    int wq = widths[lq];
    int pq = q - starts[lq];
    float rx = ((pq % wq) + 0.5f) / (float)wq;
    float ry = ((pq / wq) + 0.5f) / (float)wq;

    size_t rowb = (size_t)(b * Q_ + q) * NOA_;
    // softmax over the 16 (level,point) logits
    size_t abase = rowb + 256 + hh * (NL_ * NP_);
    float lg[16];
    float mx = -1e30f;
#pragma unroll
    for (int j = 0; j < 16; ++j) { lg[j] = offaw[abase + j]; mx = fmaxf(mx, lg[j]); }
    float s = 0.0f;
#pragma unroll
    for (int j = 0; j < 16; ++j) { lg[j] = expf(lg[j] - mx); s += lg[j]; }
    float inv = 1.0f / s;

    size_t obase = rowb + hh * (NL_ * NP_ * 2);
    float acc = 0.0f;
#pragma unroll
    for (int lvl = 0; lvl < 4; ++lvl) {
        int W = widths[lvl];
        float Wf = (float)W;
        size_t vbase = ((size_t)b * Q_ + starts[lvl]) * D_ + hh * DH_ + lane;
#pragma unroll
        for (int pp = 0; pp < 4; ++pp) {
            float ox = offaw[obase + (lvl * NP_ + pp) * 2 + 0];
            float oy = offaw[obase + (lvl * NP_ + pp) * 2 + 1];
            // loc = ref + off/wh ; sample coords (align_corners=False)
            float xs = (rx + ox / Wf) * Wf - 0.5f;
            float ys = (ry + oy / Wf) * Wf - 0.5f;
            float x0f = floorf(xs), y0f = floorf(ys);
            float wx = xs - x0f, wy = ys - y0f;
            int x0 = (int)x0f, y0 = (int)y0f;
            float c00 = 0.f, c10 = 0.f, c01 = 0.f, c11 = 0.f;
            bool xa = (unsigned)x0       < (unsigned)W;
            bool xb = (unsigned)(x0 + 1) < (unsigned)W;
            bool ya = (unsigned)y0       < (unsigned)W;
            bool yb = (unsigned)(y0 + 1) < (unsigned)W;
            if (xa && ya) c00 = v[vbase + (size_t)(y0 * W + x0) * D_];
            if (xb && ya) c10 = v[vbase + (size_t)(y0 * W + x0 + 1) * D_];
            if (xa && yb) c01 = v[vbase + (size_t)((y0 + 1) * W + x0) * D_];
            if (xb && yb) c11 = v[vbase + (size_t)((y0 + 1) * W + x0 + 1) * D_];
            float bil = (c00 * (1.f - wx) + c10 * wx) * (1.f - wy)
                      + (c01 * (1.f - wx) + c11 * wx) * wy;
            acc += lg[lvl * 4 + pp] * inv * bil;
        }
    }
    attn[(size_t)(b * Q_ + q) * D_ + hh * DH_ + lane] = (_Float16)acc;
}

// ============================================================
// x = LayerNorm(x + y) * g + be ; also emit f16 copy. One block per row.
// ============================================================
__global__ void __launch_bounds__(256) add_layernorm(
    const float* __restrict__ xin, const float* __restrict__ y,
    const float* __restrict__ g, const float* __restrict__ be,
    float* __restrict__ xout, _Float16* __restrict__ xh)
{
    __shared__ float red[256];
    int row = blockIdx.x;
    int t = threadIdx.x;
    size_t idx = (size_t)row * D_ + t;
    float val = xin[idx] + y[idx];
    red[t] = val; __syncthreads();
    for (int s = 128; s > 0; s >>= 1) { if (t < s) red[t] += red[t + s]; __syncthreads(); }
    float mean = red[0] * (1.0f / D_);
    __syncthreads();
    float dv = val - mean;
    red[t] = dv * dv; __syncthreads();
    for (int s = 128; s > 0; s >>= 1) { if (t < s) red[t] += red[t + s]; __syncthreads(); }
    float var = red[0] * (1.0f / D_);
    float o = dv * rsqrtf(var + 1e-5f) * g[t] + be[t];
    xout[idx] = o;
    xh[idx]   = (_Float16)o;
}

// ============================================================
// finalize: copy x to d_out and append spatial_shapes/level_start_index
// ============================================================
__global__ void finalize(const float* __restrict__ x, float* __restrict__ out,
                         int total, int out_size)
{
    int i = blockIdx.x * blockDim.x + threadIdx.x;
    if (i < total) out[i] = x[i];
    if (blockIdx.x == 0 && threadIdx.x < 12 && (total + 12) <= out_size) {
        const float tail[12] = {128.f, 128.f, 64.f, 64.f, 32.f, 32.f, 16.f, 16.f,
                                0.f, 16384.f, 20480.f, 21504.f};
        out[total + threadIdx.x] = tail[threadIdx.x];
    }
}

// ============================================================
extern "C" void kernel_launch(void* const* d_in, const int* in_sizes, int n_in,
                              void* d_out, int out_size, void* d_ws, size_t ws_size,
                              hipStream_t stream)
{
    (void)in_sizes; (void)n_in; (void)ws_size;

    const float* src[4] = {(const float*)d_in[0], (const float*)d_in[2],
                           (const float*)d_in[4], (const float*)d_in[6]};
    const float* pos[4] = {(const float*)d_in[1], (const float*)d_in[3],
                           (const float*)d_in[5], (const float*)d_in[7]};
    const float* lev   = (const float*)d_in[8];
    const float* W_off = (const float*)d_in[9];   const float* b_off = (const float*)d_in[10];
    const float* W_aw  = (const float*)d_in[11];  const float* b_aw  = (const float*)d_in[12];
    const float* W_v   = (const float*)d_in[13];  const float* b_v   = (const float*)d_in[14];
    const float* W_out = (const float*)d_in[15];  const float* b_out = (const float*)d_in[16];
    const float* g1    = (const float*)d_in[17];  const float* be1   = (const float*)d_in[18];
    const float* W1    = (const float*)d_in[19];  const float* b1    = (const float*)d_in[20];
    const float* W2    = (const float*)d_in[21];  const float* b2    = (const float*)d_in[22];
    const float* g2    = (const float*)d_in[23];  const float* be2   = (const float*)d_in[24];

    // ---- workspace carve-out (256B aligned) ----
    char* ws = (char*)d_ws;
    size_t cur = 0;
    auto alloc = [&](size_t bytes) -> void* {
        void* p = ws + cur;
        cur += (bytes + 255) & ~(size_t)255;
        return p;
    };
    const int M = M_;
    float*    x      = (float*)   alloc((size_t)M * D_ * 4);
    float*    posb   = (float*)   alloc((size_t)M * D_ * 4);
    _Float16* xh     = (_Float16*)alloc((size_t)M * D_ * 2);
    _Float16* qh     = (_Float16*)alloc((size_t)M * D_ * 2);
    float*    offaw  = (float*)   alloc((size_t)M * NOA_ * 4);
    float*    vbuf   = (float*)   alloc((size_t)M * D_ * 4);
    _Float16* attnh  = (_Float16*)alloc((size_t)M * D_ * 2);
    float*    ybuf   = (float*)   alloc((size_t)M * D_ * 4);
    _Float16* hbuf   = (_Float16*)alloc((size_t)M * DFF_ * 2);
    _Float16* wtoa   = (_Float16*)alloc((size_t)NLAYERS_ * NOA_ * D_ * 2);
    float*    boa    = (float*)   alloc((size_t)NLAYERS_ * NOA_ * 4);
    _Float16* wtv    = (_Float16*)alloc((size_t)NLAYERS_ * D_ * D_ * 2);
    _Float16* wtout  = (_Float16*)alloc((size_t)NLAYERS_ * D_ * D_ * 2);
    _Float16* wt1    = (_Float16*)alloc((size_t)NLAYERS_ * D_ * DFF_ * 2);
    _Float16* wt2    = (_Float16*)alloc((size_t)NLAYERS_ * DFF_ * D_ * 2);

    // ---- one-time weight transpose/convert (f32 [K,N] -> f16 [N,K]) ----
    auto launch_wt = [&](const float* W, _Float16* Wt, int K, int N,
                         long dstLayerElems, int rowOff) {
        long total = (long)NLAYERS_ * K * N;
        int blk = (int)((total + 255) / 256);
        convert_wt<<<blk, 256, 0, stream>>>(W, Wt, K, N, dstLayerElems, rowOff);
    };
    launch_wt(W_off, wtoa,  D_, 256,  (long)NOA_ * D_, 0);    // fused cols 0..255
    launch_wt(W_aw,  wtoa,  D_, 128,  (long)NOA_ * D_, 256);  // fused cols 256..383
    launch_wt(W_v,   wtv,   D_, D_,   (long)D_ * D_,   0);
    launch_wt(W_out, wtout, D_, D_,   (long)D_ * D_,   0);
    launch_wt(W1,    wt1,   D_, DFF_, (long)D_ * DFF_, 0);
    launch_wt(W2,    wt2,   DFF_, D_, (long)DFF_ * D_, 0);
    concat_bias<<<(NLAYERS_ * NOA_ + 255) / 256, 256, 0, stream>>>(b_off, b_aw, boa);

    const int totalXD = M * D_;
    pack_inputs<<<(totalXD + 255) / 256, 256, 0, stream>>>(
        src[0], pos[0], src[1], pos[1], src[2], pos[2], src[3], pos[3],
        lev, x, posb, xh);

    auto gemm = [&](const _Float16* A, const _Float16* Bt, const float* bias,
                    float* Cf, _Float16* Ch, int N, int K, int relu) {
        dim3 grid(M / 128, N / 128);
        gemm_wmma_f16<<<grid, 256, 0, stream>>>(A, Bt, bias, Cf, Ch, M, N, K, relu);
    };

    for (int l = 0; l < NLAYERS_; ++l) {
        // q = x + pos (f16)
        add_to_f16<<<(totalXD + 255) / 256, 256, 0, stream>>>(x, posb, qh, totalXD);
        // fused sampling-offsets + attention logits, and values
        gemm(qh, wtoa + (size_t)l * NOA_ * D_, boa + (size_t)l * NOA_, offaw, nullptr, NOA_, D_, 0);
        gemm(xh, wtv  + (size_t)l * D_ * D_,   b_v + (size_t)l * D_,   vbuf,  nullptr, D_,  D_, 0);
        // deformable bilinear sampling + softmax-weighted sum
        deform_sample<<<(M * NH_) / 4, 128, 0, stream>>>(offaw, vbuf, attnh);
        // output projection + residual layernorm
        gemm(attnh, wtout + (size_t)l * D_ * D_, b_out + (size_t)l * D_, ybuf, nullptr, D_, D_, 0);
        add_layernorm<<<M, 256, 0, stream>>>(x, ybuf, g1 + (size_t)l * D_, be1 + (size_t)l * D_, x, xh);
        // FFN
        gemm(xh,   wt1 + (size_t)l * D_ * DFF_, b1 + (size_t)l * DFF_, nullptr, hbuf, DFF_, D_, 1);
        gemm(hbuf, wt2 + (size_t)l * DFF_ * D_, b2 + (size_t)l * D_,   ybuf, nullptr, D_, DFF_, 0);
        add_layernorm<<<M, 256, 0, stream>>>(x, ybuf, g2 + (size_t)l * D_, be2 + (size_t)l * D_, x, xh);
    }

    finalize<<<(totalXD + 255) / 256, 256, 0, stream>>>(x, (float*)d_out, totalXD, out_size);
}
